// MobileVitBlockWithMoE_4956392260093
// MI455X (gfx1250) — compile-verified, matrix operand order checked by reading, not run on hardware
//
#include <hip/hip_runtime.h>
#include <hip/hip_bf16.h>
#include <math.h>

// ---------------- constants -----------------
#define B_     32
#define CIN_   128
#define COUT_  128
#define TD_    192
#define H_     64
#define W_     64
#define NTOK   (B_ * H_ * W_)   // 131072 tokens
#define E_     4
#define MLPH_  384
#define SEQ_   64                // tokens per sequence (NP)
#define NSEQ_  2048              // B*PA
#define HEADS_ 4
#define HD_    48
#define DEPTH_ 2
#define FOUT_  (B_ * COUT_ * H_ * W_) // 16777216

typedef _Float16 half_t;
typedef half_t v16h __attribute__((ext_vector_type(16)));
typedef half_t v8h  __attribute__((ext_vector_type(8)));
typedef float  v8f  __attribute__((ext_vector_type(8)));

// ---------------- WMMA helpers -----------------
// CDNA5 16-bit fragment (ISA 7.12.2): lane (l&15) selects the row; lanes 16-31
// use K base 8. Each lane's 16 halfs are two contiguous 8-half (16B) runs at
// K = base+0..7 and base+16..23 -> two ds_load_b128 per fragment.
__device__ __forceinline__ v16h frag_ld(const half_t* p) {
  v8h lo = *(const v8h*)(p);
  v8h hi = *(const v8h*)(p + 16);
  return __builtin_shufflevector(lo, hi, 0, 1, 2, 3, 4, 5, 6, 7,
                                 8, 9, 10, 11, 12, 13, 14, 15);
}
// works for A tiles ([m][K] row-major) and transposed-B tiles ([n][K] row-major)
__device__ __forceinline__ v16h frag_row(const half_t* p, int ld, int r0, int k0, int lane) {
  return frag_ld(p + (size_t)(r0 + (lane & 15)) * ld + k0 + ((lane >> 4) << 3));
}

__device__ __forceinline__ v8f wmma16(v16h a, v16h b, v8f c) {
  return __builtin_amdgcn_wmma_f32_16x16x32_f16(false, a, false, b, (short)0, c, false, false);
}

__device__ __forceinline__ void zero8(v8f* a, int n) {
  for (int t = 0; t < n; ++t)
#pragma unroll
    for (int i = 0; i < 8; ++i) a[t][i] = 0.f;
}

// stage a [ncols][32] transposed-B tile from a [N][K]-layout f16 weight panel
__device__ __forceinline__ void stage_bt(half_t* Bs, const half_t* src, int ldk,
                                         int k0, int ncols, int tid) {
  const uint32_t* s = (const uint32_t*)src;
  uint32_t* d = (uint32_t*)Bs;
  const int ldw = ldk >> 1, kw0 = k0 >> 1;
  for (int idx = tid; idx < ncols * 16; idx += 256) {
    int n = idx >> 4, kw = idx & 15;
    d[idx] = s[(size_t)n * ldw + kw0 + kw];
  }
}

__device__ __forceinline__ float silu_f(float v) { return v / (1.f + __expf(-v)); }
__device__ __forceinline__ float gelu_f(float v) { return 0.5f * v * (1.f + erff(v * 0.70710678118654752f)); }

// ---------------- weight conversion: f32 -> f16 [N][K] panels -----------------
__global__ void cvt_bt(const float* __restrict__ src, half_t* __restrict__ dst,
                       int K, int N, int src_kn) {
  int i = blockIdx.x * 256 + threadIdx.x;
  if (i >= N * K) return;
  int n = i / K, k = i % K;
  dst[i] = (half_t)(src_kn ? src[(size_t)k * N + n] : src[i]);
}

// ---------------- implicit-GEMM 3x3 conv + BN + SiLU -----------------
// one block per image row (b,y): M=64 x positions, N=128 out channels, K=(C0+C1)*9
__global__ __launch_bounds__(256)
void conv3x3_wmma(const float* __restrict__ in0, int C0,
                  const float* __restrict__ in1, int C1,
                  const half_t* __restrict__ wpk,   // [128][K] f16 (BT layout)
                  const float* __restrict__ bng, const float* __restrict__ bnb,
                  float* __restrict__ out, int out_nchw) {
  __shared__ __align__(16) half_t As[64 * 32];
  __shared__ __align__(16) half_t Bs[128 * 32];
  const int Ktot = (C0 + C1) * 9;
  const int by = blockIdx.x;
  const int b = by >> 6, y = by & 63;
  const int tid = threadIdx.x, wid = tid >> 5, lane = tid & 31;
  const int m0 = (wid & 3) * 16;
  const int n0 = (wid >> 2) * 64;
  v8f acc[4]; zero8(acc, 4);

  for (int k0 = 0; k0 < Ktot; k0 += 32) {
    for (int idx = tid; idx < 64 * 32; idx += 256) {
      int row = idx >> 5, kk = idx & 31;
      int k = k0 + kk;
      float v = 0.f;
      int ci = k / 9, r9 = k % 9;
      int yy = y + r9 / 3 - 1, xx = row + r9 % 3 - 1;
      if (yy >= 0 && yy < H_ && xx >= 0 && xx < W_) {
        if (ci < C0) v = in0[(((size_t)b * C0 + ci) * H_ + yy) * W_ + xx];
        else         v = in1[(((size_t)b * C1 + (ci - C0)) * H_ + yy) * W_ + xx];
      }
      As[row * 32 + kk] = (half_t)v;
    }
    stage_bt(Bs, wpk, Ktot, k0, 128, tid);
    __syncthreads();
    v16h a = frag_row(As, 32, m0, 0, lane);
#pragma unroll
    for (int nt = 0; nt < 4; ++nt)
      acc[nt] = wmma16(a, frag_row(Bs, 32, n0 + nt * 16, 0, lane), acc[nt]);
    __syncthreads();
  }
#pragma unroll
  for (int nt = 0; nt < 4; ++nt)
#pragma unroll
    for (int i = 0; i < 8; ++i) {
      int row = m0 + i + ((lane >> 4) << 3);   // x position
      int col = n0 + nt * 16 + (lane & 15);    // out channel
      float v = bng[col] * acc[nt][i] + bnb[col];
      v = silu_f(v);
      if (out_nchw) out[(((size_t)b * COUT_ + col) * H_ + y) * W_ + row] = v;
      else          out[((size_t)(b * 4096 + y * 64 + row)) * COUT_ + col] = v;
    }
}

// ---------------- 1x1: NHWC conv1 output (M=131072,K=128) x W[192][128] -> token order -----------------
__global__ __launch_bounds__(256)
void gemm_1x1_tok(const float* __restrict__ A, const half_t* __restrict__ Bw,
                  float* __restrict__ tok) {
  __shared__ __align__(16) half_t As[64 * 32];
  __shared__ __align__(16) half_t Bs[192 * 32];
  const int s0 = blockIdx.x * 64;
  const int tid = threadIdx.x, wid = tid >> 5, lane = tid & 31;
  const int m0 = (wid & 3) * 16;
  const int n0 = (wid >> 2) * 96;
  v8f acc[6]; zero8(acc, 6);
  for (int k0 = 0; k0 < 128; k0 += 32) {
    for (int idx = tid; idx < 64 * 32; idx += 256) {
      int row = idx >> 5, kk = idx & 31;
      As[idx] = (half_t)A[(size_t)(s0 + row) * 128 + k0 + kk];
    }
    stage_bt(Bs, Bw, 128, k0, 192, tid);
    __syncthreads();
    v16h a = frag_row(As, 32, m0, 0, lane);
#pragma unroll
    for (int nt = 0; nt < 6; ++nt)
      acc[nt] = wmma16(a, frag_row(Bs, 32, n0 + nt * 16, 0, lane), acc[nt]);
    __syncthreads();
  }
#pragma unroll
  for (int nt = 0; nt < 6; ++nt)
#pragma unroll
    for (int i = 0; i < 8; ++i) {
      int row = m0 + i + ((lane >> 4) << 3);
      int col = n0 + nt * 16 + (lane & 15);
      int s = s0 + row;
      int b = s >> 12, y = (s >> 6) & 63, x = s & 63;
      int r = b * 4096 + ((y & 7) * 8 + (x & 7)) * 64 + ((y >> 3) * 8 + (x >> 3));
      tok[(size_t)r * 192 + col] = acc[nt][i];
    }
}

// ---------------- fused MoE: gating + 2 GEMMs per expert -----------------
__global__ __launch_bounds__(256)
void moe_wmma(const float* __restrict__ tok, const float* __restrict__ gate_w,
              const int* __restrict__ taskp,
              const half_t* __restrict__ w1, const float* __restrict__ b1,
              const half_t* __restrict__ w2, const float* __restrict__ b2,
              float* __restrict__ outp, float* __restrict__ stats) {
  __shared__ __align__(16) half_t Tt[64 * 192];
  __shared__ __align__(16) half_t He[64 * 192];
  __shared__ __align__(16) half_t Bs[192 * 32];
  __shared__ float  Gw[192 * 4];
  __shared__ float  Dense[64 * 4];
  const int t0 = blockIdx.x * 64;
  const int tid = threadIdx.x, wid = tid >> 5, lane = tid & 31;
  const int task = *taskp;
  for (int idx = tid; idx < 64 * 192; idx += 256)
    Tt[idx] = (half_t)tok[(size_t)t0 * 192 + idx];
  for (int idx = tid; idx < 192 * 4; idx += 256)
    Gw[idx] = gate_w[(size_t)task * 768 + idx];
  __syncthreads();
  if (tid < 64) {
    float lg[4] = {0.f, 0.f, 0.f, 0.f};
    for (int d = 0; d < 192; ++d) {
      float x = (float)Tt[tid * 192 + d];
      for (int e = 0; e < 4; ++e) lg[e] += x * Gw[d * 4 + e];
    }
    float mx = lg[0];
    for (int e = 1; e < 4; ++e) mx = fmaxf(mx, lg[e]);
    float p[4], sum = 0.f;
    for (int e = 0; e < 4; ++e) { p[e] = __expf(lg[e] - mx); sum += p[e]; }
    for (int e = 0; e < 4; ++e) p[e] /= sum;
    int i1 = 0;
    for (int e = 1; e < 4; ++e) if (p[e] > p[i1]) i1 = e;
    int i2 = (i1 == 0) ? 1 : 0;
    for (int e = 0; e < 4; ++e) if (e != i1 && p[e] > p[i2]) i2 = e;
    float tv = p[i1] + p[i2];
    float dns[4] = {0.f, 0.f, 0.f, 0.f};
    dns[i1] = p[i1] / tv; dns[i2] = p[i2] / tv;
    for (int e = 0; e < 4; ++e) {
      Dense[tid * 4 + e] = dns[e];
      atomicAdd(&stats[e], p[e]);
      atomicAdd(&stats[4 + e], dns[e] > 0.f ? 1.f : 0.f);
    }
  }
  __syncthreads();
  const int m0 = (wid & 3) * 16;
  const int n0 = (wid >> 2) * 96;
  v8f oacc[6]; zero8(oacc, 6);
  for (int e = 0; e < 4; ++e) {
    v8f acc[6]; zero8(acc, 6);
    for (int k0 = 0; k0 < 192; k0 += 32) {
      stage_bt(Bs, w1 + (size_t)e * 36864, 192, k0, 192, tid);
      __syncthreads();
      v16h a = frag_row(Tt, 192, m0, k0, lane);
#pragma unroll
      for (int nt = 0; nt < 6; ++nt)
        acc[nt] = wmma16(a, frag_row(Bs, 32, n0 + nt * 16, 0, lane), acc[nt]);
      __syncthreads();
    }
#pragma unroll
    for (int nt = 0; nt < 6; ++nt)
#pragma unroll
      for (int i = 0; i < 8; ++i) {
        int row = m0 + i + ((lane >> 4) << 3);
        int col = n0 + nt * 16 + (lane & 15);
        He[row * 192 + col] = (half_t)gelu_f(acc[nt][i] + b1[e * 192 + col]);
      }
    __syncthreads();
    zero8(acc, 6);
    for (int k0 = 0; k0 < 192; k0 += 32) {
      stage_bt(Bs, w2 + (size_t)e * 36864, 192, k0, 192, tid);
      __syncthreads();
      v16h a = frag_row(He, 192, m0, k0, lane);
#pragma unroll
      for (int nt = 0; nt < 6; ++nt)
        acc[nt] = wmma16(a, frag_row(Bs, 32, n0 + nt * 16, 0, lane), acc[nt]);
      __syncthreads();
    }
#pragma unroll
    for (int nt = 0; nt < 6; ++nt)
#pragma unroll
      for (int i = 0; i < 8; ++i) {
        int row = m0 + i + ((lane >> 4) << 3);
        int col = n0 + nt * 16 + (lane & 15);
        oacc[nt][i] += Dense[row * 4 + e] * (acc[nt][i] + b2[e * 192 + col]);
      }
    __syncthreads();
  }
#pragma unroll
  for (int nt = 0; nt < 6; ++nt)
#pragma unroll
    for (int i = 0; i < 8; ++i) {
      int row = m0 + i + ((lane >> 4) << 3);
      int col = n0 + nt * 16 + (lane & 15);
      outp[(size_t)(t0 + row) * 192 + col] = oacc[nt][i];
    }
}

__global__ void moe_loss_k(const float* __restrict__ stats, float* __restrict__ out) {
  if (threadIdx.x == 0 && blockIdx.x == 0) {
    float s = 0.f;
    for (int e = 0; e < 4; ++e)
      s += (stats[e] / (float)NTOK) * (stats[4 + e] / (float)NTOK);
    out[0] = 4.f * s;
  }
}

// ---------------- fused ViT encoder layer (one block per sequence) -----------------
// dynamic LDS layout (bytes):
//  yb   f16 64x192          @ 0       (24576)
//  qkvS f16 64x512 q/k slots@ 24576   (65536)  [row][(q|k)*256 + head*64 + d], d 48..63 zero
//        (aliased by hb f16 64x384 in MLP phase)
//  vT   f16 4x48x64         @ 90112   (24576)  [head][d][s]
//  scb  f32 2x64x64         @ 114688  (32768)
//  attb f16 2x64x64         @ 147456  (16384)
//  Bs   f16 192x32          @ 163840  (12288)  -> total 176128 bytes
__global__ __launch_bounds__(256)
void vit_layer(float* __restrict__ t,
               const float* __restrict__ ln1g, const float* __restrict__ ln1b,
               const half_t* __restrict__ qkvw, const float* __restrict__ qkvbias,
               const half_t* __restrict__ projw, const float* __restrict__ projbias,
               const float* __restrict__ ln2g, const float* __restrict__ ln2b,
               const half_t* __restrict__ fc1w, const float* __restrict__ fc1bias,
               const half_t* __restrict__ fc2w, const float* __restrict__ fc2bias) {
  extern __shared__ __align__(16) char smem[];
  half_t* yb   = (half_t*)(smem);
  half_t* qkvS = (half_t*)(smem + 24576);
  half_t* hb   = qkvS;
  half_t* vT   = (half_t*)(smem + 90112);
  float*  scb  = (float*)(smem + 114688);
  half_t* attb = (half_t*)(smem + 147456);
  half_t* Bs   = (half_t*)(smem + 163840);

  const int n = blockIdx.x;
  float* tn = t + (size_t)n * SEQ_ * TD_;
  const int tid = threadIdx.x, wid = tid >> 5, lane = tid & 31;
  const int m0 = (wid & 3) * 16;

  // zero q/k slots (covers the d=48..63 padding used by the padded-K WMMAs)
  for (int idx = tid; idx < 64 * 512 / 2; idx += 256)
    ((uint32_t*)qkvS)[idx] = 0u;

  // ---- LN1 ----
  if (tid < 64) {
    const float* r = tn + tid * 192;
    float m = 0.f;
    for (int d = 0; d < 192; ++d) m += r[d];
    m *= (1.f / 192.f);
    float v = 0.f;
    for (int d = 0; d < 192; ++d) { float dd = r[d] - m; v += dd * dd; }
    v *= (1.f / 192.f);
    float rs = rsqrtf(v + 1e-5f);
    for (int d = 0; d < 192; ++d)
      yb[tid * 192 + d] = (half_t)((r[d] - m) * rs * ln1g[d] + ln1b[d]);
  }
  __syncthreads();

  // ---- QKV (N=576 in 3 chunks of 192), scatter into padded q/k slots + vT ----
  for (int nc = 0; nc < 3; ++nc) {
    const int nbase = nc * 192;
    const int n0 = (wid >> 2) * 96;
    v8f acc[6]; zero8(acc, 6);
    for (int k0 = 0; k0 < 192; k0 += 32) {
      stage_bt(Bs, qkvw + (size_t)nbase * 192, 192, k0, 192, tid);
      __syncthreads();
      v16h a = frag_row(yb, 192, m0, k0, lane);
#pragma unroll
      for (int nt = 0; nt < 6; ++nt)
        acc[nt] = wmma16(a, frag_row(Bs, 32, n0 + nt * 16, 0, lane), acc[nt]);
      __syncthreads();
    }
#pragma unroll
    for (int nt = 0; nt < 6; ++nt)
#pragma unroll
      for (int i = 0; i < 8; ++i) {
        int row = m0 + i + ((lane >> 4) << 3);
        int gc = nbase + n0 + nt * 16 + (lane & 15);
        float val = acc[nt][i] + qkvbias[gc];
        int which = gc / 192, rem = gc % 192;
        int head = rem / 48, d = rem % 48;
        if (which == 2) vT[((head * 48) + d) * 64 + row] = (half_t)val;
        else            qkvS[row * 512 + which * 256 + head * 64 + d] = (half_t)val;
      }
  }
  __syncthreads();

  // ---- attention: two head-groups of 2 heads ----
  const float scale = 0.14433756729740645f; // 48^-0.5
  for (int hg = 0; hg < 2; ++hg) {
    const int lh = wid >> 2;        // local head 0/1
    const int head = hg * 2 + lh;
    // scores = q @ k^T  (K = hd zero-padded 48->64); qkvS is [row][d]:
    // A rows are queries, B rows (transposed layout) are keys.
    {
      v8f acc[4]; zero8(acc, 4);
      for (int k0 = 0; k0 < 64; k0 += 32) {
        v16h a = frag_row(qkvS + head * 64, 512, m0, k0, lane);
#pragma unroll
        for (int nt = 0; nt < 4; ++nt)
          acc[nt] = wmma16(a, frag_row(qkvS + 256 + head * 64, 512, nt * 16, k0, lane), acc[nt]);
      }
#pragma unroll
      for (int nt = 0; nt < 4; ++nt)
#pragma unroll
        for (int i = 0; i < 8; ++i) {
          int q = m0 + i + ((lane >> 4) << 3);
          int kk = nt * 16 + (lane & 15);
          scb[lh * 4096 + q * 64 + kk] = acc[nt][i] * scale;
        }
    }
    __syncthreads();
    // softmax per (head, query)
    if (tid < 128) {
      int l2 = tid >> 6, q = tid & 63;
      float* row = scb + l2 * 4096 + q * 64;
      float mx = -1e30f;
      for (int k = 0; k < 64; ++k) mx = fmaxf(mx, row[k]);
      float s = 0.f;
      for (int k = 0; k < 64; ++k) s += __expf(row[k] - mx);
      float inv = 1.f / s;
      for (int k = 0; k < 64; ++k)
        attb[(l2 * 64 + q) * 64 + k] = (half_t)(__expf(row[k] - mx) * inv);
    }
    __syncthreads();
    // y = att @ v : vT[head][d][s] is the transposed-B tile ([n=d][k=s])
    {
      v8f acc[3]; zero8(acc, 3);
      for (int k0 = 0; k0 < 64; k0 += 32) {
        v16h a = frag_row(attb + lh * 4096, 64, m0, k0, lane);
#pragma unroll
        for (int nt = 0; nt < 3; ++nt)
          acc[nt] = wmma16(a, frag_row(vT + head * 48 * 64, 64, nt * 16, k0, lane), acc[nt]);
      }
#pragma unroll
      for (int nt = 0; nt < 3; ++nt)
#pragma unroll
        for (int i = 0; i < 8; ++i) {
          int row = m0 + i + ((lane >> 4) << 3);
          int col = head * 48 + nt * 16 + (lane & 15);
          yb[row * 192 + col] = (half_t)acc[nt][i];
        }
    }
    __syncthreads();
  }

  // ---- proj + residual ----
  {
    const int n0 = (wid >> 2) * 96;
    v8f acc[6]; zero8(acc, 6);
    for (int k0 = 0; k0 < 192; k0 += 32) {
      stage_bt(Bs, projw, 192, k0, 192, tid);
      __syncthreads();
      v16h a = frag_row(yb, 192, m0, k0, lane);
#pragma unroll
      for (int nt = 0; nt < 6; ++nt)
        acc[nt] = wmma16(a, frag_row(Bs, 32, n0 + nt * 16, 0, lane), acc[nt]);
      __syncthreads();
    }
#pragma unroll
    for (int nt = 0; nt < 6; ++nt)
#pragma unroll
      for (int i = 0; i < 8; ++i) {
        int row = m0 + i + ((lane >> 4) << 3);
        int col = n0 + nt * 16 + (lane & 15);
        tn[row * 192 + col] += acc[nt][i] + projbias[col];
      }
  }
  __syncthreads();

  // ---- LN2 ----
  if (tid < 64) {
    const float* r = tn + tid * 192;
    float m = 0.f;
    for (int d = 0; d < 192; ++d) m += r[d];
    m *= (1.f / 192.f);
    float v = 0.f;
    for (int d = 0; d < 192; ++d) { float dd = r[d] - m; v += dd * dd; }
    v *= (1.f / 192.f);
    float rs = rsqrtf(v + 1e-5f);
    for (int d = 0; d < 192; ++d)
      yb[tid * 192 + d] = (half_t)((r[d] - m) * rs * ln2g[d] + ln2b[d]);
  }
  __syncthreads();

  // ---- FC1 (N=384 in 2 chunks) + GELU ----
  for (int nc = 0; nc < 2; ++nc) {
    const int nbase = nc * 192;
    const int n0 = (wid >> 2) * 96;
    v8f acc[6]; zero8(acc, 6);
    for (int k0 = 0; k0 < 192; k0 += 32) {
      stage_bt(Bs, fc1w + (size_t)nbase * 192, 192, k0, 192, tid);
      __syncthreads();
      v16h a = frag_row(yb, 192, m0, k0, lane);
#pragma unroll
      for (int nt = 0; nt < 6; ++nt)
        acc[nt] = wmma16(a, frag_row(Bs, 32, n0 + nt * 16, 0, lane), acc[nt]);
      __syncthreads();
    }
#pragma unroll
    for (int nt = 0; nt < 6; ++nt)
#pragma unroll
      for (int i = 0; i < 8; ++i) {
        int row = m0 + i + ((lane >> 4) << 3);
        int gc = nbase + n0 + nt * 16 + (lane & 15);
        hb[row * 384 + gc] = (half_t)gelu_f(acc[nt][i] + fc1bias[gc]);
      }
    __syncthreads();
  }

  // ---- FC2 (K=384) + residual ----
  {
    const int n0 = (wid >> 2) * 96;
    v8f acc[6]; zero8(acc, 6);
    for (int k0 = 0; k0 < 384; k0 += 32) {
      stage_bt(Bs, fc2w, 384, k0, 192, tid);
      __syncthreads();
      v16h a = frag_row(hb, 384, m0, k0, lane);
#pragma unroll
      for (int nt = 0; nt < 6; ++nt)
        acc[nt] = wmma16(a, frag_row(Bs, 32, n0 + nt * 16, 0, lane), acc[nt]);
      __syncthreads();
    }
#pragma unroll
    for (int nt = 0; nt < 6; ++nt)
#pragma unroll
      for (int i = 0; i < 8; ++i) {
        int row = m0 + i + ((lane >> 4) << 3);
        int col = n0 + nt * 16 + (lane & 15);
        tn[row * 192 + col] += acc[nt][i] + fc2bias[col];
      }
  }
}

// ---------------- final LayerNorm over tokens -----------------
__global__ __launch_bounds__(256)
void final_ln(const float* __restrict__ t, const float* __restrict__ g,
              const float* __restrict__ b, float* __restrict__ out) {
  int r = blockIdx.x * 256 + threadIdx.x;
  if (r >= NTOK) return;
  const float* row = t + (size_t)r * 192;
  float m = 0.f;
  for (int d = 0; d < 192; ++d) m += row[d];
  m *= (1.f / 192.f);
  float v = 0.f;
  for (int d = 0; d < 192; ++d) { float dd = row[d] - m; v += dd * dd; }
  v *= (1.f / 192.f);
  float rs = rsqrtf(v + 1e-5f);
  float* o = out + (size_t)r * 192;
  for (int d = 0; d < 192; ++d) o[d] = (row[d] - m) * rs * g[d] + b[d];
}

// ---------------- token x w_proj + BN + SiLU -> NCHW p -----------------
__global__ __launch_bounds__(256)
void gemm_proj(const float* __restrict__ A, const half_t* __restrict__ Bw,
               const float* __restrict__ g, const float* __restrict__ bb,
               float* __restrict__ p) {
  __shared__ __align__(16) half_t As[64 * 32];
  __shared__ __align__(16) half_t Bs[128 * 32];
  const int r0 = blockIdx.x * 64;
  const int tid = threadIdx.x, wid = tid >> 5, lane = tid & 31;
  const int m0 = (wid & 3) * 16;
  const int n0 = (wid >> 2) * 64;
  v8f acc[4]; zero8(acc, 4);
  for (int k0 = 0; k0 < 192; k0 += 32) {
    for (int idx = tid; idx < 64 * 32; idx += 256) {
      int row = idx >> 5, kk = idx & 31;
      As[idx] = (half_t)A[(size_t)(r0 + row) * 192 + k0 + kk];
    }
    stage_bt(Bs, Bw, 192, k0, 128, tid);
    __syncthreads();
    v16h a = frag_row(As, 32, m0, 0, lane);
#pragma unroll
    for (int nt = 0; nt < 4; ++nt)
      acc[nt] = wmma16(a, frag_row(Bs, 32, n0 + nt * 16, 0, lane), acc[nt]);
    __syncthreads();
  }
#pragma unroll
  for (int nt = 0; nt < 4; ++nt)
#pragma unroll
    for (int i = 0; i < 8; ++i) {
      int lrow = m0 + i + ((lane >> 4) << 3);
      int col = n0 + nt * 16 + (lane & 15);
      int r = r0 + lrow;
      int b = r >> 12, pa = (r >> 6) & 63, np = r & 63;
      int y = (np >> 3) * 8 + (pa >> 3);
      int x = (np & 7) * 8 + (pa & 7);
      float v = silu_f(g[col] * acc[nt][i] + bb[col]);
      p[(((size_t)b * COUT_ + col) * H_ + y) * W_ + x] = v;
    }
}

// ---------------- host launch -----------------
extern "C" void kernel_launch(void* const* d_in, const int* in_sizes, int n_in,
                              void* d_out, int out_size, void* d_ws, size_t ws_size,
                              hipStream_t stream) {
  (void)in_sizes; (void)n_in; (void)out_size; (void)ws_size;
  const float* x      = (const float*)d_in[0];
  const float* w_kxk  = (const float*)d_in[1];
  const float* bn1_g  = (const float*)d_in[2];
  const float* bn1_b  = (const float*)d_in[3];
  const float* w_1x1  = (const float*)d_in[4];
  const float* gate_w = (const float*)d_in[5];
  const float* moe_w1 = (const float*)d_in[6];
  const float* moe_b1 = (const float*)d_in[7];
  const float* moe_w2 = (const float*)d_in[8];
  const float* moe_b2 = (const float*)d_in[9];
  const float* ln1_g  = (const float*)d_in[10];
  const float* ln1_b  = (const float*)d_in[11];
  const float* qkv_w  = (const float*)d_in[12];
  const float* qkv_b  = (const float*)d_in[13];
  const float* proj_w = (const float*)d_in[14];
  const float* proj_b = (const float*)d_in[15];
  const float* ln2_g  = (const float*)d_in[16];
  const float* ln2_b  = (const float*)d_in[17];
  const float* fc1_w  = (const float*)d_in[18];
  const float* fc1_b  = (const float*)d_in[19];
  const float* fc2_w  = (const float*)d_in[20];
  const float* fc2_b  = (const float*)d_in[21];
  const float* nrm_g  = (const float*)d_in[22];
  const float* nrm_b  = (const float*)d_in[23];
  const float* w_proj = (const float*)d_in[24];
  const float* bn2_g  = (const float*)d_in[25];
  const float* bn2_b  = (const float*)d_in[26];
  const float* w_fus  = (const float*)d_in[27];
  const float* bn3_g  = (const float*)d_in[28];
  const float* bn3_b  = (const float*)d_in[29];
  const int*   taskp  = (const int*)d_in[30];

  char* ws = (char*)d_ws;
  size_t off = 0;
  auto alloc = [&](size_t bytes) -> void* {
    void* p = ws + off;
    off = (off + bytes + 255) & ~(size_t)255;
    return p;
  };
  float*  c1out  = (float*)alloc((size_t)NTOK * 128 * 4);  // NHWC conv1 output
  float*  tok    = (float*)alloc((size_t)NTOK * 192 * 4);  // tokens (token order)
  float*  moe_t  = (float*)alloc((size_t)NTOK * 192 * 4);  // residual stream
  float*  lnout  = (float*)alloc((size_t)NTOK * 192 * 4);  // final LN output
  float*  pbuf   = (float*)alloc((size_t)NTOK * 128 * 4);  // NCHW projected
  // all f16 weight panels are [N][K] ("BT") layout
  half_t* wk1    = (half_t*)alloc((size_t)128 * 1152 * 2);
  half_t* wkf    = (half_t*)alloc((size_t)128 * 2304 * 2);
  half_t* w1x1h  = (half_t*)alloc((size_t)192 * 128 * 2);
  half_t* moe1h  = (half_t*)alloc((size_t)4 * 192 * 192 * 2);
  half_t* moe2h  = (half_t*)alloc((size_t)4 * 192 * 192 * 2);
  half_t* qkvwh  = (half_t*)alloc((size_t)2 * 576 * 192 * 2);
  half_t* projwh = (half_t*)alloc((size_t)2 * 192 * 192 * 2);
  half_t* fc1wh  = (half_t*)alloc((size_t)2 * 384 * 192 * 2);
  half_t* fc2wh  = (half_t*)alloc((size_t)2 * 192 * 384 * 2);
  half_t* wprojh = (half_t*)alloc((size_t)128 * 192 * 2);
  float*  stats  = (float*)alloc(8 * 4);

  hipMemsetAsync(stats, 0, 8 * sizeof(float), stream);

  // allow >64KB dynamic LDS for the fused layer kernel (176128 bytes)
  hipFuncSetAttribute((const void*)vit_layer,
                      hipFuncAttributeMaxDynamicSharedMemorySize, 176128);

  auto nblk = [](int n) { return (n + 255) / 256; };
  // weight packing into [N][K] f16 panels
  cvt_bt<<<nblk(128 * 1152), 256, 0, stream>>>(w_kxk, wk1, 1152, 128, 0);   // (O,I,3,3) is already [N][K]
  cvt_bt<<<nblk(128 * 2304), 256, 0, stream>>>(w_fus, wkf, 2304, 128, 0);
  cvt_bt<<<nblk(192 * 128), 256, 0, stream>>>(w_1x1, w1x1h, 128, 192, 0);   // (TD,CIN) is [N][K]
  for (int e = 0; e < E_; ++e) {
    cvt_bt<<<nblk(192 * 192), 256, 0, stream>>>(moe_w1 + (size_t)e * 36864,
                                                moe1h + (size_t)e * 36864, 192, 192, 1);
    cvt_bt<<<nblk(192 * 192), 256, 0, stream>>>(moe_w2 + (size_t)e * 36864,
                                                moe2h + (size_t)e * 36864, 192, 192, 1);
  }
  for (int l = 0; l < DEPTH_; ++l) {
    cvt_bt<<<nblk(576 * 192), 256, 0, stream>>>(qkv_w + (size_t)l * 110592,
                                                qkvwh + (size_t)l * 110592, 192, 576, 1);
    cvt_bt<<<nblk(192 * 192), 256, 0, stream>>>(proj_w + (size_t)l * 36864,
                                                projwh + (size_t)l * 36864, 192, 192, 1);
    cvt_bt<<<nblk(384 * 192), 256, 0, stream>>>(fc1_w + (size_t)l * 73728,
                                                fc1wh + (size_t)l * 73728, 192, 384, 1);
    cvt_bt<<<nblk(192 * 384), 256, 0, stream>>>(fc2_w + (size_t)l * 73728,
                                                fc2wh + (size_t)l * 73728, 384, 192, 1);
  }
  cvt_bt<<<nblk(128 * 192), 256, 0, stream>>>(w_proj, wprojh, 192, 128, 0);  // (COUT,TD) is [N][K]

  // conv1 + BN + SiLU -> NHWC
  conv3x3_wmma<<<B_ * H_, 256, 0, stream>>>(x, 128, nullptr, 0, wk1,
                                            bn1_g, bn1_b, c1out, 0);
  // 1x1 -> tokens (token order)
  gemm_1x1_tok<<<NTOK / 64, 256, 0, stream>>>(c1out, w1x1h, tok);
  // fused MoE
  moe_wmma<<<NTOK / 64, 256, 0, stream>>>(tok, gate_w, taskp, moe1h, moe_b1,
                                          moe2h, moe_b2, moe_t, stats);
  moe_loss_k<<<1, 32, 0, stream>>>(stats, (float*)d_out + FOUT_);
  // transformer encoder, DEPTH=2
  for (int l = 0; l < DEPTH_; ++l) {
    vit_layer<<<NSEQ_, 256, 176128, stream>>>(
        moe_t,
        ln1_g + l * 192, ln1_b + l * 192,
        qkvwh + (size_t)l * 110592, qkv_b + l * 576,
        projwh + (size_t)l * 36864, proj_b + l * 192,
        ln2_g + l * 192, ln2_b + l * 192,
        fc1wh + (size_t)l * 73728, fc1_b + l * 384,
        fc2wh + (size_t)l * 73728, fc2_b + l * 192);
  }
  // final LN
  final_ln<<<NTOK / 256, 256, 0, stream>>>(moe_t, nrm_g, nrm_b, lnout);
  // project to COUT, BN + SiLU, scatter NCHW
  gemm_proj<<<NTOK / 64, 256, 0, stream>>>(lnout, wprojh, bn2_g, bn2_b, pbuf);
  // fusion conv over concat(x, p) + BN + SiLU -> d_out NCHW
  conv3x3_wmma<<<B_ * H_, 256, 0, stream>>>(x, 128, pbuf, 128, wkf,
                                            bn3_g, bn3_b, (float*)d_out, 1);
}
